// Emb_MLP_Layers_18279380811821
// MI455X (gfx1250) — matrix operand order repaired and checked by reading, not compiled, not run on hardware
//
#include <hip/hip_runtime.h>
#include <hip/hip_bf16.h>

typedef __bf16 bf16_t;
typedef __attribute__((ext_vector_type(16))) __bf16 v16bf;
typedef __attribute__((ext_vector_type(8)))  __bf16 v8bf;
typedef __attribute__((ext_vector_type(4)))  __bf16 v4bf;
typedef __attribute__((ext_vector_type(8)))  float  v8f;

#define N_NODES 100000
#define N_EDGES 1600000
#define N_REL   16
#define D_EMB   128
#define D_HID   64
#define D_LAB   50
#define D_OUTF  242
#define D_OUTFP 256

#define BROW    72      // LDS row stride (bf16 elems) for 64-wide B panel: 144B, 16B-aligned
#define KMAX    256     // largest K any GEMM uses

union AFrag { v16bf v; v8bf h[2]; };

// ---------------------------------------------------------------------------
// Weight pack: f32 [B, K, Ns] (row-major) -> bf16 [B, Kp, Np], zero padded.
// ---------------------------------------------------------------------------
__global__ void pack_weights(const float* __restrict__ W, bf16_t* __restrict__ out,
                             int B, int K, int Kp, int Ns, int Np) {
    size_t tid = (size_t)blockIdx.x * blockDim.x + threadIdx.x;
    size_t total = (size_t)B * Kp * Np;
    if (tid >= total) return;
    int n = (int)(tid % Np);
    size_t t = tid / Np;
    int k = (int)(t % Kp);
    int b = (int)(t / Kp);
    float v = (k < K && n < Ns) ? W[((size_t)b * K + k) * Ns + n] : 0.0f;
    out[((size_t)b * Kp + k) * Np + n] = (bf16_t)v;
}

// f32 -> bf16 elementwise
__global__ void cvt_bf16(const float* __restrict__ in, bf16_t* __restrict__ out, size_t n) {
    size_t tid = (size_t)blockIdx.x * blockDim.x + threadIdx.x;
    if (tid < n) out[tid] = (bf16_t)in[tid];
}

// ---------------------------------------------------------------------------
// Edge degree count per (relation, dst) segment.
// ---------------------------------------------------------------------------
__global__ void count_edges(const int* __restrict__ dst, const int* __restrict__ et,
                            float* __restrict__ cnt, int E) {
    size_t e = (size_t)blockIdx.x * blockDim.x + threadIdx.x;
    if (e >= (size_t)E) return;
    atomicAdd(&cnt[(size_t)et[e] * N_NODES + dst[e]], 1.0f);
}

// Scatter-add x[src] (bf16) into f32 sums[(et*N+dst)*D + c], 4 elems/thread.
__global__ void scatter_edges(const bf16_t* __restrict__ x, const int* __restrict__ src,
                              const int* __restrict__ dst, const int* __restrict__ et,
                              float* __restrict__ sums, int E, int D) {
    size_t tid = (size_t)blockIdx.x * blockDim.x + threadIdx.x;
    int per = D >> 2;
    if (tid >= (size_t)E * per) return;
    int e = (int)(tid / per);
    int c = (int)(tid % per) * 4;
    int s = src[e];
    size_t seg = (size_t)et[e] * N_NODES + dst[e];
    v4bf xv = *(const v4bf*)(x + (size_t)s * D + c);
    float* o = sums + seg * D + c;
    atomicAdd(&o[0], (float)xv[0]);
    atomicAdd(&o[1], (float)xv[1]);
    atomicAdd(&o[2], (float)xv[2]);
    atomicAdd(&o[3], (float)xv[3]);
}

// mean = sums / max(cnt,1), packed to bf16 row-major [R*N, D].
__global__ void mean_pack(const float* __restrict__ sums, const float* __restrict__ cnt,
                          bf16_t* __restrict__ mean, size_t total4, int D) {
    size_t tid = (size_t)blockIdx.x * blockDim.x + threadIdx.x;
    if (tid >= total4) return;
    size_t idx = tid * 4;
    size_t seg = idx / D;
    float inv = 1.0f / fmaxf(cnt[seg], 1.0f);
    v4bf m;
    m[0] = (bf16_t)(sums[idx + 0] * inv);
    m[1] = (bf16_t)(sums[idx + 1] * inv);
    m[2] = (bf16_t)(sums[idx + 2] * inv);
    m[3] = (bf16_t)(sums[idx + 3] * inv);
    *(v4bf*)(mean + idx) = m;
}

// ---------------------------------------------------------------------------
// Unified WMMA GEMM with LDS-staged B panels:
//   out = act( sum_{r<Rrel} mean_r @ wpack_r  +  xin @ rootp  + bias )
// Rrel==0 -> plain dense layer (xin @ rootp + bias).
// A sources: row-major bf16 [M, K] (K multiple of 32).
// B sources: row-major bf16 [K, Np] (Np multiple of 64, zero-padded).
// Block = 4 waves; each wave owns a 16(M) x 64(N) strip (4 f32 accumulators).
// Per source: all 128 threads stage B[K, nb..nb+63] into LDS, then waves feed
// WMMA from LDS (ds_load) while A fragments stream from global (loadcnt).
// ACT: 0 = sigmoid, 1 = relu.  PADZ: write 0 for cols in [Nvalid, Np).
// ---------------------------------------------------------------------------
template <int ACT, typename OT, bool PADZ>
__global__ __launch_bounds__(128) void gemm_wmma(
    const bf16_t* __restrict__ xin, const bf16_t* __restrict__ mean,
    const bf16_t* __restrict__ wpack, const bf16_t* __restrict__ rootp,
    const float* __restrict__ bias, OT* __restrict__ out,
    int M, int K, int Np, int Nvalid, int ldo, int Rrel) {
    __shared__ bf16_t smem[KMAX * BROW];   // 36 KB of the WGP's 320 KB LDS

    int tid  = threadIdx.x;
    int wave = tid >> 5;
    int lane = tid & 31;
    int tileM = blockIdx.x * 4 + wave;
    bool valid = (tileM * 16 < M);         // wave-uniform; M % 16 == 0
    int m0 = tileM * 16;
    int nb = blockIdx.y * 64;
    int hh  = lane >> 4;                   // which K-half this lane holds (A frag)
    int l15 = lane & 15;

    v8f acc[4];
#pragma unroll
    for (int j = 0; j < 4; ++j) acc[j] = (v8f){0.f, 0.f, 0.f, 0.f, 0.f, 0.f, 0.f, 0.f};

    for (int s = 0; s <= Rrel; ++s) {
        const bf16_t* A = (s < Rrel) ? (mean + (size_t)s * M * K) : xin;
        const bf16_t* B = (s < Rrel) ? (wpack + (size_t)s * K * Np) : rootp;

        // ---- cooperative stage of B[K, nb:nb+64] into LDS ----
        __syncthreads();                   // previous source fully consumed
        {
            int chunks = K * 8;            // 8 x 16B chunks per 64-wide row
            for (int i = tid; i < chunks; i += 128) {
                int row = i >> 3, c = i & 7;
                v8bf vv = *(const v8bf*)(B + (size_t)row * Np + nb + c * 8);
                *(v8bf*)(&smem[row * BROW + c * 8]) = vv;
            }
        }
        __syncthreads();

        if (valid) {
            const bf16_t* arow = A + (size_t)(m0 + l15) * K + hh * 8;
            const bf16_t* blane = smem + l15;   // lane = N column within panel
            for (int k0 = 0; k0 < K; k0 += 32) {
                AFrag a;
                a.h[0] = *(const v8bf*)(arow + k0);        // K = k0 + hh*8 .. +7
                a.h[1] = *(const v8bf*)(arow + k0 + 16);   // K = k0+16+hh*8 .. +7
                v16bf b0, b1, b2, b3;
#pragma unroll
                for (int e = 0; e < 16; ++e) {
                    const bf16_t* bl = blane + (k0 + hh * 8 + (e & 7) + (e >> 3) * 16) * BROW;
                    b0[e] = bl[0];
                    b1[e] = bl[16];
                    b2[e] = bl[32];
                    b3[e] = bl[48];
                }
                acc[0] = __builtin_amdgcn_wmma_f32_16x16x32_bf16(
                    false, a.v, false, b0, (short)0, acc[0], false, false);
                acc[1] = __builtin_amdgcn_wmma_f32_16x16x32_bf16(
                    false, a.v, false, b1, (short)0, acc[1], false, false);
                acc[2] = __builtin_amdgcn_wmma_f32_16x16x32_bf16(
                    false, a.v, false, b2, (short)0, acc[2], false, false);
                acc[3] = __builtin_amdgcn_wmma_f32_16x16x32_bf16(
                    false, a.v, false, b3, (short)0, acc[3], false, false);
            }
        }
    }

    if (!valid) return;
#pragma unroll
    for (int j = 0; j < 4; ++j) {
        int col = nb + j * 16 + l15;
        float bv = (col < Nvalid) ? bias[col] : 0.0f;
#pragma unroll
        for (int v = 0; v < 8; ++v) {
            int row = m0 + hh * 8 + v;
            float x = acc[j][v] + bv;
            float y;
            if (ACT == 0) y = 1.0f / (1.0f + __expf(-x));
            else          y = fmaxf(x, 0.0f);
            if (col < Nvalid)      out[(size_t)row * ldo + col] = (OT)y;
            else if (PADZ)         out[(size_t)row * ldo + col] = (OT)0.0f;
        }
    }
}

// ---------------------------------------------------------------------------
extern "C" void kernel_launch(void* const* d_in, const int* in_sizes, int n_in,
                              void* d_out, int out_size, void* d_ws, size_t ws_size,
                              hipStream_t stream) {
    const float* emb    = (const float*)d_in[0];
    const float* lin1_w = (const float*)d_in[1];
    const float* lin1_b = (const float*)d_in[2];
    const float* lin2_w = (const float*)d_in[3];
    const float* lin2_b = (const float*)d_in[4];
    const float* w1     = (const float*)d_in[5];
    const float* root1  = (const float*)d_in[6];
    const float* b1     = (const float*)d_in[7];
    const float* w2     = (const float*)d_in[8];
    const float* root2  = (const float*)d_in[9];
    const float* b2     = (const float*)d_in[10];
    const int*   eidx   = (const int*)d_in[11];
    const int*   etype  = (const int*)d_in[12];
    const int* src = eidx;
    const int* dst = eidx + N_EDGES;
    float* outp = (float*)d_out;

    // ---- workspace carving (256B aligned) ----
    char* base = (char*)d_ws;
    size_t off = 0;
    auto carve = [&](size_t bytes) -> char* {
        char* p = base + off;
        off = (off + bytes + 255) & ~(size_t)255;
        return p;
    };
    bf16_t* embBF = (bf16_t*)carve((size_t)N_NODES * D_EMB * 2);
    bf16_t* x1    = (bf16_t*)carve((size_t)N_NODES * D_OUTFP * 2);
    bf16_t* x2    = (bf16_t*)carve((size_t)N_NODES * D_EMB * 2);
    bf16_t* hbuf  = (bf16_t*)carve((size_t)N_NODES * D_HID * 2);
    float*  cnt   = (float*)carve((size_t)N_REL * N_NODES * 4);
    bf16_t* l1p   = (bf16_t*)carve((size_t)D_EMB * D_OUTFP * 2);
    bf16_t* l2p   = (bf16_t*)carve((size_t)D_OUTFP * D_EMB * 2);
    bf16_t* r1p   = (bf16_t*)carve((size_t)D_EMB * D_HID * 2);
    bf16_t* w1p   = (bf16_t*)carve((size_t)N_REL * D_EMB * D_HID * 2);
    bf16_t* r2p   = (bf16_t*)carve((size_t)D_HID * D_HID * 2);   // 64x64 (50 padded)
    bf16_t* w2p   = (bf16_t*)carve((size_t)N_REL * D_HID * D_HID * 2);
    float*  sums  = (float*)carve((size_t)N_REL * N_NODES * D_EMB * 4);   // reused for conv2
    bf16_t* mean  = (bf16_t*)carve((size_t)N_REL * N_NODES * D_EMB * 2);  // reused for conv2

    const int TPB = 256;
    auto blocks = [](size_t n, int t) { return (unsigned)((n + t - 1) / t); };

    // ---- weight packing (tiny, once per call) ----
    pack_weights<<<blocks((size_t)D_EMB * D_OUTFP, TPB), TPB, 0, stream>>>(lin1_w, l1p, 1, D_EMB, D_EMB, D_OUTF, D_OUTFP);
    pack_weights<<<blocks((size_t)D_OUTFP * D_EMB, TPB), TPB, 0, stream>>>(lin2_w, l2p, 1, D_OUTF, D_OUTFP, D_EMB, D_EMB);
    pack_weights<<<blocks((size_t)D_EMB * D_HID, TPB), TPB, 0, stream>>>(root1, r1p, 1, D_EMB, D_EMB, D_HID, D_HID);
    pack_weights<<<blocks((size_t)N_REL * D_EMB * D_HID, TPB), TPB, 0, stream>>>(w1, w1p, N_REL, D_EMB, D_EMB, D_HID, D_HID);
    pack_weights<<<blocks((size_t)D_HID * D_HID, TPB), TPB, 0, stream>>>(root2, r2p, 1, D_HID, D_HID, D_LAB, D_HID);
    pack_weights<<<blocks((size_t)N_REL * D_HID * D_HID, TPB), TPB, 0, stream>>>(w2, w2p, N_REL, D_HID, D_HID, D_LAB, D_HID);

    // ---- emb -> bf16 ----
    cvt_bf16<<<blocks((size_t)N_NODES * D_EMB, TPB), TPB, 0, stream>>>(emb, embBF, (size_t)N_NODES * D_EMB);

    // ---- MLP layer 1: x1 = sigmoid(emb @ lin1_w + b), bf16 [N, 256] padded ----
    {
        dim3 g((N_NODES / 16 + 3) / 4, D_OUTFP / 64);
        gemm_wmma<0, bf16_t, true><<<g, 128, 0, stream>>>(
            embBF, nullptr, nullptr, l1p, lin1_b, x1,
            N_NODES, D_EMB, D_OUTFP, D_OUTF, D_OUTFP, 0);
    }
    // ---- MLP layer 2: x2 = sigmoid(x1 @ lin2_w + b), bf16 [N, 128] ----
    {
        dim3 g((N_NODES / 16 + 3) / 4, D_EMB / 64);
        gemm_wmma<0, bf16_t, false><<<g, 128, 0, stream>>>(
            x1, nullptr, nullptr, l2p, lin2_b, x2,
            N_NODES, D_OUTFP, D_EMB, D_EMB, D_EMB, 0);
    }

    // ---- edge degree counts (shared by both convs) ----
    hipMemsetAsync(cnt, 0, (size_t)N_REL * N_NODES * 4, stream);
    count_edges<<<blocks((size_t)N_EDGES, TPB), TPB, 0, stream>>>(dst, etype, cnt, N_EDGES);

    // ---- conv1 aggregation: sums[R*N,128] += x2[src] ----
    hipMemsetAsync(sums, 0, (size_t)N_REL * N_NODES * D_EMB * 4, stream);
    scatter_edges<<<blocks((size_t)N_EDGES * (D_EMB / 4), TPB), TPB, 0, stream>>>(
        x2, src, dst, etype, sums, N_EDGES, D_EMB);
    mean_pack<<<blocks((size_t)N_REL * N_NODES * (D_EMB / 4), TPB), TPB, 0, stream>>>(
        sums, cnt, mean, (size_t)N_REL * N_NODES * (D_EMB / 4), D_EMB);

    // ---- conv1 GEMM: h = relu(sum_r mean_r @ W1_r + x2 @ root1 + b1), bf16 [N,64] ----
    {
        dim3 g((N_NODES / 16 + 3) / 4, D_HID / 64);
        gemm_wmma<1, bf16_t, false><<<g, 128, 0, stream>>>(
            x2, mean, w1p, r1p, b1, hbuf,
            N_NODES, D_EMB, D_HID, D_HID, D_HID, N_REL);
    }

    // ---- conv2 aggregation: reuse sums/mean regions with D=64 ----
    hipMemsetAsync(sums, 0, (size_t)N_REL * N_NODES * D_HID * 4, stream);
    scatter_edges<<<blocks((size_t)N_EDGES * (D_HID / 4), TPB), TPB, 0, stream>>>(
        hbuf, src, dst, etype, sums, N_EDGES, D_HID);
    mean_pack<<<blocks((size_t)N_REL * N_NODES * (D_HID / 4), TPB), TPB, 0, stream>>>(
        sums, cnt, mean, (size_t)N_REL * N_NODES * (D_HID / 4), D_HID);

    // ---- conv2 GEMM: out = sigmoid(sum_r mean_r @ W2_r + h @ root2 + b2), f32 [N,50] ----
    {
        dim3 g((N_NODES / 16 + 3) / 4, 1);
        gemm_wmma<0, float, false><<<g, 128, 0, stream>>>(
            hbuf, mean, w2p, r2p, b2, outp,
            N_NODES, D_HID, D_HID, D_LAB, D_LAB, N_REL);
    }
}
